// TradingNLNN_6365141532766
// MI455X (gfx1250) — matrix operand order repaired and checked by reading, not compiled, not run on hardware
//
#include <hip/hip_runtime.h>
#include <hip/hip_bf16.h>
#include <math.h>

// ---------------- types ----------------
typedef __attribute__((ext_vector_type(16))) __bf16        v16bf;
typedef __attribute__((ext_vector_type(8)))  float         v8f;
typedef __attribute__((ext_vector_type(4)))  float         v4f;
typedef __attribute__((ext_vector_type(4)))  unsigned int  v4u;

#define Bb   2048
#define Tt   128
#define Hh   512
#define INN  5
#define OUTT 3
#define LL   2

// ---------------- prologue: row-normalize rec_w, fold rec_s, cast to bf16 ----
__global__ void prep_rec(const float* __restrict__ rec_w,
                         const float* __restrict__ rec_s,
                         __bf16* __restrict__ wout) {
  const int row = blockIdx.x;               // 0 .. L*H-1  (flat [l][j])
  const float* wr = rec_w + (size_t)row * Hh;
  __shared__ float red[64];
  float ss = 0.f;
  for (int e = threadIdx.x; e < Hh; e += 64) { float v = wr[e]; ss += v * v; }
  red[threadIdx.x] = ss;
  __syncthreads();
  for (int s = 32; s > 0; s >>= 1) {
    if (threadIdx.x < s) red[threadIdx.x] += red[threadIdx.x + s];
    __syncthreads();
  }
  const float scale = rec_s[row] / (sqrtf(red[0]) + 1e-8f);
  __bf16* wo = wout + (size_t)row * Hh;
  for (int e = threadIdx.x; e < Hh; e += 64) wo[e] = (__bf16)(wr[e] * scale);
}

// ---------------- prologue: small weights (in_w, out_w, sigmoid(alpha)) -----
__global__ void prep_small(const float* __restrict__ in_w,  const float* __restrict__ in_s,
                           const float* __restrict__ out_w, const float* __restrict__ out_s,
                           const float* __restrict__ alpha,
                           float* __restrict__ inw, float* __restrict__ outw,
                           float* __restrict__ asig) {
  const int tid = threadIdx.x;
  if (tid < Hh) {                       // in_w rows [H][IN]
    const float* w = in_w + tid * INN;
    float ss = 0.f;
    #pragma unroll
    for (int i = 0; i < INN; ++i) ss += w[i] * w[i];
    const float sc = in_s[tid] / (sqrtf(ss) + 1e-8f);
    #pragma unroll
    for (int i = 0; i < INN; ++i) inw[tid * INN + i] = w[i] * sc;
  }
  if (tid < OUTT) {                     // out_w rows [OUT][H]
    const float* w = out_w + tid * Hh;
    float ss = 0.f;
    for (int j = 0; j < Hh; ++j) ss += w[j] * w[j];
    const float sc = out_s[tid] / (sqrtf(ss) + 1e-8f);
    for (int j = 0; j < Hh; ++j) outw[tid * Hh + j] = w[j] * sc;
  }
  if (tid < LL * Hh) asig[tid] = 1.f / (1.f + expf(-alpha[tid]));
}

// ---------------- one recurrent timestep for one layer slab -----------------
// grid: (B/16, L); block: 256 threads = 8 wave32
__global__ __launch_bounds__(256)
void step_kernel(const float* __restrict__ x,
                 const float* __restrict__ ln_w, const float* __restrict__ ln_b,
                 const float* __restrict__ inw,            // [H][IN]  normalized*in_s
                 const __bf16* __restrict__ recw,          // [L][H][H] normalized*rec_s, bf16
                 const float* __restrict__ outw,           // [OUT][H] normalized*out_s
                 const float* __restrict__ asig,           // [L][H] sigmoid(alpha)
                 float* __restrict__ h,                    // [L][B][H] fp32 state
                 float* __restrict__ out,                  // outs [B][T][OUT] ++ h_final
                 int t) {
  __shared__ float  s_ip[16][Hh];       // input_proj, then h_unnorm, then slerp vec
  __shared__ __bf16 sA[16 * Hh];        // bf16 copy of the block's 16 h rows
  __shared__ float  red[3][16][16];
  __shared__ float  rowsc[16][4];       // theta, 1/sin, 1/|h_old|, 1/|h_new|
  __shared__ float  s_fin[16];

  const int tid  = threadIdx.x;
  const int b0   = blockIdx.x * 16;
  const int l    = blockIdx.y;
  const int lane = tid & 31;
  const int wv   = tid >> 5;            // wave 0..7 -> 64 output columns each
  const int rm   = tid >> 4;            // row 0..15 for reduction phases
  const int rt   = tid & 15;            // 16 threads per row, 32 contiguous cols each

  const float* hrow = h + ((size_t)l * Bb + b0 + rm) * Hh;

  // ---- stage h rows into LDS as bf16 (A operand, shared by all 8 waves) ----
  {
    const v4f* hv = (const v4f*)(hrow + rt * 32);
    #pragma unroll
    for (int c = 0; c < 8; ++c) {
      v4f p = hv[c];
      const int j = rt * 32 + c * 4;
      sA[rm * Hh + j + 0] = (__bf16)p.x;
      sA[rm * Hh + j + 1] = (__bf16)p.y;
      sA[rm * Hh + j + 2] = (__bf16)p.z;
      sA[rm * Hh + j + 3] = (__bf16)p.w;
    }
  }

  // ---- input projection: layernorm(x_t) @ in_w^T into s_ip ----
  {
    const float* xr = x + ((size_t)(b0 + rm) * Tt + t) * INN;
    float xv[INN]; float mu = 0.f;
    #pragma unroll
    for (int i = 0; i < INN; ++i) { xv[i] = xr[i]; mu += xv[i]; }
    mu *= (1.f / INN);
    float var = 0.f;
    #pragma unroll
    for (int i = 0; i < INN; ++i) { float d = xv[i] - mu; var += d * d; }
    var *= (1.f / INN);
    const float inv = rsqrtf(var + 1e-5f);
    float xn[INN];
    #pragma unroll
    for (int i = 0; i < INN; ++i) xn[i] = (xv[i] - mu) * inv * ln_w[i] + ln_b[i];
    for (int jj = 0; jj < 32; ++jj) {
      const int j = rt * 32 + jj;
      const float* wr = inw + j * INN;
      float s = 0.f;
      #pragma unroll
      for (int i = 0; i < INN; ++i) s += xn[i] * wr[i];
      s_ip[rm][j] = s;
    }
  }
  __syncthreads();

  // ---- recurrent GEMM: rp[16x512] = h_tile[16x512] @ W^T via WMMA bf16 ----
  v8f acc[4];
  {
    v8f zz = {0.f, 0.f, 0.f, 0.f, 0.f, 0.f, 0.f, 0.f};
    acc[0] = zz; acc[1] = zz; acc[2] = zz; acc[3] = zz;
  }
  const int half = lane >> 4;
  const int mA   = lane & 15;
  const __bf16* wbase = recw + (size_t)l * Hh * Hh;
  const int nb = wv * 64;

  #pragma unroll 2
  for (int kt = 0; kt < 16; ++kt) {
    // A fragment (16x32 bf16) straight from LDS:
    // lane half 0 -> K kt*32+[0..7],[16..23]; half 1 -> +8
    const __bf16* ap = sA + mA * Hh + kt * 32 + half * 8;
    union UA { v4u u[2]; v16bf v; } au;
    au.u[0] = *(const v4u*)ap;
    au.u[1] = *(const v4u*)(ap + 16);

    // prefetch next K-block of this wave's weight columns
    if (kt + 1 < 16)
      __builtin_prefetch(wbase + (size_t)(nb + (lane & 15)) * Hh + (kt + 1) * 32, 0, 3);

    #pragma unroll
    for (int t4 = 0; t4 < 4; ++t4) {
      // B fragment (32x16 bf16): lane n holds col j = nb+t4*16+n, K [0..7],[16..23] (+8 for half 1)
      const __bf16* wr = wbase + (size_t)(nb + t4 * 16 + (lane & 15)) * Hh
                               + kt * 32 + half * 8;
      union UB { v4u u[2]; v16bf v; } bu;
      bu.u[0] = *(const v4u*)wr;
      bu.u[1] = *(const v4u*)(wr + 16);
      acc[t4] = __builtin_amdgcn_wmma_f32_16x16x32_bf16(
          false, au.v, false, bu.v, (short)0, acc[t4], false, false);
    }
  }

  // ---- h_unnorm = input_proj + rp, scatter C frags into s_ip ----
  #pragma unroll
  for (int t4 = 0; t4 < 4; ++t4) {
    #pragma unroll
    for (int i = 0; i < 8; ++i) {
      const int m = i + half * 8;               // C layout: vgpr i, lane>=16 -> M+8
      const int n = nb + t4 * 16 + (lane & 15);
      s_ip[m][n] += acc[t4][i];
    }
  }
  __syncthreads();

  // ---- row reductions: |h_new|^2, dot(h_new,h_old), |h_old|^2 ----
  {
    float sn = 0.f, sd = 0.f, so = 0.f;
    for (int jj = 0; jj < 32; ++jj) {
      const int j = rt * 32 + jj;
      const float hn = s_ip[rm][j];
      const float ho = hrow[j];
      sn += hn * hn; sd += hn * ho; so += ho * ho;
    }
    red[0][rm][rt] = sn; red[1][rm][rt] = sd; red[2][rm][rt] = so;
  }
  __syncthreads();
  if (rt == 0) {
    float a0 = 0.f, a1 = 0.f, a2 = 0.f;
    for (int k = 0; k < 16; ++k) { a0 += red[0][rm][k]; a1 += red[1][rm][k]; a2 += red[2][rm][k]; }
    const float invn = 1.f / fmaxf(sqrtf(a0), 1e-8f);
    const float invo = 1.f / fmaxf(sqrtf(a2), 1e-8f);
    float dot = a1 * invn * invo;
    dot = fminf(fmaxf(dot, -1.f + 1e-8f), 1.f - 1e-8f);
    const float theta = acosf(dot);
    rowsc[rm][0] = theta;
    rowsc[rm][1] = 1.f / (sinf(theta) + 1e-8f);
    rowsc[rm][2] = invo;
    rowsc[rm][3] = invn;
  }
  __syncthreads();

  // ---- slerp combine ----
  {
    const float theta  = rowsc[rm][0];
    const float invsin = rowsc[rm][1];
    const float invo   = rowsc[rm][2];
    const float invn   = rowsc[rm][3];
    const float* as    = asig + l * Hh;
    float s2 = 0.f;
    for (int jj = 0; jj < 32; ++jj) {
      const int j = rt * 32 + jj;
      const float hn = s_ip[rm][j] * invn;
      const float ho = hrow[j] * invo;
      const float a  = as[j];
      const float ct = __sinf((1.f - a) * theta) * invsin;
      const float cn = __sinf(a * theta) * invsin;
      const float v  = ct * ho + cn * hn;
      s_ip[rm][j] = v;
      s2 += v * v;
    }
    red[0][rm][rt] = s2;
  }
  __syncthreads();
  if (rt == 0) {
    float a0 = 0.f;
    for (int k = 0; k < 16; ++k) a0 += red[0][rm][k];
    s_fin[rm] = 1.f / fmaxf(sqrtf(a0), 1e-8f);
  }
  __syncthreads();

  // ---- final normalize, state update, output projection ----
  {
    const float invf = s_fin[rm];
    float* hrow_w = h + ((size_t)l * Bb + b0 + rm) * Hh;
    float o0 = 0.f, o1 = 0.f, o2 = 0.f;
    for (int jj = 0; jj < 32; ++jj) {
      const int j = rt * 32 + jj;
      const float v = s_ip[rm][j] * invf;
      hrow_w[j] = v;
      if (l == LL - 1) {
        o0 += v * outw[j];
        o1 += v * outw[Hh + j];
        o2 += v * outw[2 * Hh + j];
      }
      if (t == Tt - 1)
        out[(size_t)Bb * Tt * OUTT + ((size_t)(l * Bb + b0 + rm)) * Hh + j] = v;
    }
    if (l == LL - 1) {
      red[0][rm][rt] = o0; red[1][rm][rt] = o1; red[2][rm][rt] = o2;
      __syncthreads();
      if (tid < 16 * OUTT) {
        const int m = tid / OUTT, o = tid % OUTT;
        float s = 0.f;
        for (int k = 0; k < 16; ++k) s += red[o][m][k];
        out[((size_t)(b0 + m) * Tt + t) * OUTT + o] = s;
      }
    }
  }
}

// ---------------- host ----------------
extern "C" void kernel_launch(void* const* d_in, const int* in_sizes, int n_in,
                              void* d_out, int out_size, void* d_ws, size_t ws_size,
                              hipStream_t stream) {
  const float* x      = (const float*)d_in[0];
  const float* h_init = (const float*)d_in[1];
  const float* ln_w   = (const float*)d_in[2];
  const float* ln_b   = (const float*)d_in[3];
  const float* in_w   = (const float*)d_in[4];
  const float* in_s   = (const float*)d_in[5];
  const float* rec_w  = (const float*)d_in[6];
  const float* rec_s  = (const float*)d_in[7];
  const float* alpha  = (const float*)d_in[8];
  const float* out_w  = (const float*)d_in[9];
  const float* out_s  = (const float*)d_in[10];
  float* out = (float*)d_out;

  char* ws = (char*)d_ws;
  float*  h    = (float*) (ws + 0);                     // L*B*H fp32   = 8 MiB
  __bf16* recw = (__bf16*)(ws + 8388608);               // L*H*H bf16   = 1 MiB
  float*  inw  = (float*) (ws + 8388608 + 1048576);     // H*IN fp32
  float*  outw = (float*) (ws + 8388608 + 1048576 + 10240);
  float*  asig = (float*) (ws + 8388608 + 1048576 + 10240 + 6144);

  hipMemcpyAsync(h, h_init, (size_t)LL * Bb * Hh * sizeof(float),
                 hipMemcpyDeviceToDevice, stream);
  prep_rec<<<LL * Hh, 64, 0, stream>>>(rec_w, rec_s, recw);
  prep_small<<<1, 1024, 0, stream>>>(in_w, in_s, out_w, out_s, alpha, inw, outw, asig);

  for (int t = 0; t < Tt; ++t) {
    step_kernel<<<dim3(Bb / 16, LL), 256, 0, stream>>>(
        x, ln_w, ln_b, inw, recw, outw, asig, h, out, t);
  }
}
